// MS_SSA_63565515980889
// MI455X (gfx1250) — matrix operand order repaired and checked by reading, not compiled, not run on hardware
//
#include <hip/hip_runtime.h>
#include <math.h>

typedef __attribute__((ext_vector_type(16))) _Float16 v16h;
typedef __attribute__((ext_vector_type(8)))  _Float16 v8h;
typedef __attribute__((ext_vector_type(8)))  float    v8f;
typedef int v4i __attribute__((vector_size(16)));

#define TS   4
#define BBATCH 8
#define DCH  256
#define NLEN 1024
#define HEADS 8
#define NSEQ 128                 // NLEN / HEADS
#define PER_T (BBATCH*DCH*NLEN)  // 2,097,152
#define TOT   (TS*PER_T)         // 8,388,608
#define VTH   1.0f
#define EPSBN 1e-5f

// ---- CDNA5 async LDS staging (guarded: falls back to plain LDS stores) ----
#if defined(__has_builtin)
#if __has_builtin(__builtin_amdgcn_global_load_async_to_lds_b128) && \
    __has_builtin(__builtin_amdgcn_s_wait_asynccnt)
#define HAVE_ASYNC_LDS 1
#endif
#endif

static __device__ __forceinline__ void stage16(const _Float16* g, _Float16* l) {
#ifdef HAVE_ASYNC_LDS
  __builtin_amdgcn_global_load_async_to_lds_b128(
      (__attribute__((address_space(1))) v4i*)(v4i*)g,
      (__attribute__((address_space(3))) v4i*)(v4i*)l, 0, 0);
#else
  *(uint4*)l = *(const uint4*)g;
#endif
}
static __device__ __forceinline__ void stage_fence() {
#ifdef HAVE_ASYNC_LDS
  __builtin_amdgcn_s_wait_asynccnt(0);
#endif
  __syncthreads();
}

// ---------------------------------------------------------------- softmax(w)
__global__ void k_softmax(const float* __restrict__ w, float* __restrict__ wsm) {
  __shared__ float red[256];
  int t = threadIdx.x;
  float v = w[t];
  red[t] = v; __syncthreads();
  for (int s = 128; s > 0; s >>= 1) { if (t < s) red[t] = fmaxf(red[t], red[t+s]); __syncthreads(); }
  float mx = red[0]; __syncthreads();
  float e = __expf(v - mx);
  red[t] = e; __syncthreads();
  for (int s = 128; s > 0; s >>= 1) { if (t < s) red[t] += red[t+s]; __syncthreads(); }
  wsm[t] = e / red[0];
}

// ---------------------------------------------------------------- rope tables
__global__ void k_rope_tables(float* __restrict__ cosT, float* __restrict__ sinT) {
  int idx = blockIdx.x * 256 + threadIdx.x;    // NSEQ*DCH = 32768
  int ns = idx >> 8, dd = idx & 255;
  float expo = (float)(dd & ~1) / 256.0f;      // 2*(dd/2)/d
  float f = (float)ns * powf(10000.0f, -expo);
  cosT[idx] = cosf(f);
  sinT[idx] = sinf(f);
}

// ---------------------------------------------------------------- PLIF (head_lif)
__global__ void k_plif(const float* __restrict__ x, const float* __restrict__ wsm,
                       const float* __restrict__ bias, float* __restrict__ s) {
  int idx = blockIdx.x * 256 + threadIdx.x;    // over PER_T
  int d = (idx >> 10) & 255;
  float wv = wsm[d], bv = bias[d];
  float v = 0.f;
  #pragma unroll
  for (int t = 0; t < TS; ++t) {
    float xt = x[(size_t)t * PER_T + idx];
    v += (xt - v) * wv + bv;
    float sp = (v >= VTH) ? 1.f : 0.f;
    s[(size_t)t * PER_T + idx] = sp;
    v = (sp > 0.f) ? 0.f : v;                  // hard reset, v_reset = 0
  }
}

// ---------------------------------------------------------------- LIF (decay_input=False)
__global__ void k_lif(const float* __restrict__ x, float* __restrict__ s) {
  int idx = blockIdx.x * 256 + threadIdx.x;
  float v = 0.f;
  #pragma unroll
  for (int t = 0; t < TS; ++t) {
    float xt = x[(size_t)t * PER_T + idx];
    v = v * 0.5f + xt;
    float sp = (v >= VTH) ? 1.f : 0.f;
    s[(size_t)t * PER_T + idx] = sp;
    v = (sp > 0.f) ? 0.f : v;
  }
}

// ---------------------------------------------------------------- fused BN-affine + LIF
__global__ void k_bn_lif(const float* __restrict__ y, const float* __restrict__ sc,
                         const float* __restrict__ sh, float* __restrict__ s) {
  int idx = blockIdx.x * 256 + threadIdx.x;
  int d = (idx >> 10) & 255;
  float a = sc[d], b = sh[d];
  float v = 0.f;
  #pragma unroll
  for (int t = 0; t < TS; ++t) {
    float xt = y[(size_t)t * PER_T + idx] * a + b;
    v = v * 0.5f + xt;
    float sp = (v >= VTH) ? 1.f : 0.f;
    s[(size_t)t * PER_T + idx] = sp;
    v = (sp > 0.f) ? 0.f : v;
  }
}

// -------------------------------------------- fused BN + LIF + (rope) -> f16 head tiles
// kv_layout=0: out[tbh][nseq][dd] (row-major; used for Q, K-dim = dd contiguous)
// kv_layout=1: out[tbh][dd][nseq] (transposed;  used for K/V, K-dim = nseq contiguous)
__global__ void k_bn_lif_rope_f16(const float* __restrict__ y, const float* __restrict__ sc,
                                  const float* __restrict__ sh, const float* __restrict__ cosT,
                                  const float* __restrict__ sinT, int do_rope, int kv_layout,
                                  _Float16* __restrict__ out) {
  int p  = blockIdx.x * 256 + threadIdx.x;     // pair over PER_T/2
  int i0 = p * 2;                              // even element in (B,D,N) slab
  int ii = i0 & (DCH * NLEN - 1);              // within one batch's (D,N) block (2^18)
  int bb = i0 >> 18;
  int d  = ii >> 10;
  int hf = ii >> 15;
  int ns = (ii >> 8) & 127;
  int dd = ii & 255;                           // even
  float a = sc[d], b = sh[d];
  float c0 = 1.f, s0 = 0.f, c1 = 1.f, s1 = 0.f;
  if (do_rope) {
    c0 = cosT[ns * 256 + dd];     s0 = sinT[ns * 256 + dd];
    c1 = cosT[ns * 256 + dd + 1]; s1 = sinT[ns * 256 + dd + 1];
  }
  float v0 = 0.f, v1 = 0.f;
  #pragma unroll
  for (int t = 0; t < TS; ++t) {
    float x0 = y[(size_t)t * PER_T + i0]     * a + b;
    float x1 = y[(size_t)t * PER_T + i0 + 1] * a + b;
    v0 = v0 * 0.5f + x0;  v1 = v1 * 0.5f + x1;
    float sp0 = (v0 >= VTH) ? 1.f : 0.f;
    float sp1 = (v1 >= VTH) ? 1.f : 0.f;
    v0 = (sp0 > 0.f) ? 0.f : v0;
    v1 = (sp1 > 0.f) ? 0.f : v1;
    float r0 = sp0, r1 = sp1;
    if (do_rope) { r0 = sp0 * c0 - sp1 * s0;  r1 = sp1 * c1 + sp0 * s1; }
    size_t base = ((size_t)((t * BBATCH + bb) * HEADS + hf)) * (NSEQ * DCH);
    if (kv_layout) {
      out[base + (size_t)dd * NSEQ + ns]       = (_Float16)r0;
      out[base + (size_t)(dd + 1) * NSEQ + ns] = (_Float16)r1;
    } else {
      out[base + (size_t)ns * DCH + dd]     = (_Float16)r0;
      out[base + (size_t)ns * DCH + dd + 1] = (_Float16)r1;
    }
  }
}

// ---------------------------------------------------------------- depthwise conv + BN stats
__global__ void k_dwconv_stats(const float* __restrict__ x, const float* __restrict__ w,
                               int K, float* __restrict__ y,
                               float* __restrict__ s1, float* __restrict__ s2) {
  __shared__ float row[NLEN + 8];
  __shared__ float red[256];
  int md = blockIdx.x;                 // m*D + d
  int d  = md & 255;
  int t  = threadIdx.x;
  int pad = (K - 1) >> 1;
  const float* xr = x + (size_t)md * NLEN;
  if (t == 0) __builtin_prefetch(x + (size_t)(md + 1) * NLEN, 0, 0);  // global_prefetch_b8
  for (int j = t; j < NLEN + K - 1; j += 256) {
    int xi = j - pad;
    row[j] = (xi >= 0 && xi < NLEN) ? xr[xi] : 0.f;
  }
  float wk[5];
  #pragma unroll
  for (int k = 0; k < 5; ++k) wk[k] = (k < K) ? w[d * K + k] : 0.f;
  __syncthreads();
  float ls = 0.f, lq = 0.f;
  #pragma unroll
  for (int c = 0; c < NLEN / 256; ++c) {
    int i = c * 256 + t;
    float acc = 0.f;
    #pragma unroll
    for (int k = 0; k < 5; ++k) if (k < K) acc += row[i + k] * wk[k];
    y[(size_t)md * NLEN + i] = acc;
    ls += acc; lq += acc * acc;
  }
  red[t] = ls; __syncthreads();
  for (int s = 128; s > 0; s >>= 1) { if (t < s) red[t] += red[t+s]; __syncthreads(); }
  if (t == 0) atomicAdd(&s1[d], red[0]);
  __syncthreads();
  red[t] = lq; __syncthreads();
  for (int s = 128; s > 0; s >>= 1) { if (t < s) red[t] += red[t+s]; __syncthreads(); }
  if (t == 0) atomicAdd(&s2[d], red[0]);
}

// ---------------------------------------------------------------- BN finalize scale/shift
__global__ void k_bn_finalize(const float* __restrict__ s1, const float* __restrict__ s2,
                              const float* __restrict__ g, const float* __restrict__ b,
                              float cnt, float* __restrict__ sc, float* __restrict__ sh) {
  int d = threadIdx.x;
  float m   = s1[d] / cnt;
  float var = s2[d] / cnt - m * m;
  float a   = g[d] * rsqrtf(var + EPSBN);
  sc[d] = a;
  sh[d] = b[d] - m * a;
}

// ---------------------------------------------------------------- BN apply (final output)
__global__ void k_bn_apply(const float* __restrict__ y, const float* __restrict__ sc,
                           const float* __restrict__ sh, float* __restrict__ out) {
  size_t idx = (size_t)blockIdx.x * 256 + threadIdx.x;   // TOT
  int d = ((int)(idx >> 10)) & 255;
  out[idx] = y[idx] * sc[d] + sh[d];
}

// ---------------------------------------------------------------- WMMA fragment loads
// All operands are stored so the WMMA K-dim is unit-stride:
//   A (16x32 f16): lane's 16 elements = two contiguous 16B chunks (ISA 7.12.2)
//   B (32x16 f16): lane's 16 elements = two contiguous 16B chunks (same as A)
static __device__ __forceinline__ v16h frag_a(const _Float16* rowp, int lane) {
  int hi = (lane >> 4) & 1;
  v8h lo = *(const v8h*)(rowp + hi * 8);        // k = hi*8 .. hi*8+7
  v8h hh = *(const v8h*)(rowp + 16 + hi * 8);   // k = 16+hi*8 .. +7
  v16h a;
  #pragma unroll
  for (int i = 0; i < 8; ++i) { a[i] = lo[i]; a[8 + i] = hh[i]; }
  return a;
}
static __device__ __forceinline__ v16h frag_b(const _Float16* colp, int lane) {
  int hi = (lane >> 4) & 1;
  v8h lo = *(const v8h*)(colp + hi * 8);        // k = hi*8 .. hi*8+7
  v8h hh = *(const v8h*)(colp + 16 + hi * 8);   // k = 16+hi*8 .. +7
  v16h b;
  #pragma unroll
  for (int i = 0; i < 8; ++i) { b[i] = lo[i]; b[8 + i] = hh[i]; }
  return b;
}

// ---------------------------------------------------------------- kv^T = (K^T V)^T * scale
// Inputs Kt/Vt: [tbh][dd][nseq] (n contiguous). Output kvT: [tbh][e][d] (d contiguous).
// One block per (t,b,h). LDS: Kt + Vt = 128 KB (dynamic).
__global__ void k_kv_wmma(const _Float16* __restrict__ Kg, const _Float16* __restrict__ Vg,
                          _Float16* __restrict__ kvT) {
  extern __shared__ _Float16 smem[];
  _Float16* Ks = smem;                 // [d][n] 256x128
  _Float16* Vs = smem + DCH * NSEQ;    // [e][n] 256x128
  int tbh = blockIdx.x;
  const _Float16* kg = Kg + (size_t)tbh * DCH * NSEQ;
  const _Float16* vg = Vg + (size_t)tbh * DCH * NSEQ;
  for (int j = threadIdx.x; j < DCH * NSEQ / 8; j += 256) {
    stage16(kg + j * 8, Ks + j * 8);
    stage16(vg + j * 8, Vs + j * 8);
  }
  stage_fence();
  int lane = threadIdx.x & 31, wave = threadIdx.x >> 5;
  int m = lane & 15;
  const float scale = 0.0883883476483f;   // 1/sqrt(128)
  for (int it = 0; it < 32; ++it) {       // 8 waves x 32 tiles = 256 (16x16) tiles
    int tile = wave * 32 + it;
    int dt = tile >> 4, et = tile & 15;
    const _Float16* arow = Ks + (size_t)(dt * 16 + m) * NSEQ;  // A[m][k]=K[n][d] (K^T)
    const _Float16* bcol = Vs + (size_t)(et * 16 + m) * NSEQ;  // B[k][n]=V[n][e]
    v8f acc = {};
    #pragma unroll
    for (int kc = 0; kc < 4; ++kc) {      // inner n = 128 = 4 * 32
      v16h a = frag_a(arow + kc * 32, lane);
      v16h b = frag_b(bcol + kc * 32, lane);
      acc = __builtin_amdgcn_wmma_f32_16x16x32_f16(false, a, false, b, (short)0, acc, false, false);
    }
    // D row = dt*16 + v + 8*hi, col = et*16 + (lane&15): store transposed, contiguous 16B
    int hi = lane >> 4;
    v8h hv;
    #pragma unroll
    for (int v = 0; v < 8; ++v) hv[v] = (_Float16)(acc[v] * scale);
    *(v8h*)(kvT + (size_t)tbh * DCH * DCH + (size_t)(et * 16 + m) * DCH + dt * 16 + hi * 8) = hv;
  }
}

// ---------------------------------------------------------------- att = Q @ kv
// Inputs: Q [tbh][nseq][dd] (d contiguous), kvT [tbh][e][d] (d contiguous).
// One block per (t,b,h). LDS: Q(64KB) + kvT(128KB) = 192 KB (dynamic).
// Writes directly into the swapaxes(2,3)+reshape (Ts,B,D,N) layout.
__global__ void k_att_wmma(const _Float16* __restrict__ Qg, const _Float16* __restrict__ KVg,
                           float* __restrict__ att) {
  extern __shared__ _Float16 smem[];
  _Float16* Qs = smem;                 // [nseq][d] 128x256
  _Float16* Cs = smem + NSEQ * DCH;    // [e][d]    256x256
  int tbh = blockIdx.x;
  const _Float16* qg = Qg  + (size_t)tbh * NSEQ * DCH;
  const _Float16* cg = KVg + (size_t)tbh * DCH * DCH;
  for (int j = threadIdx.x; j < NSEQ * DCH / 8; j += 256) stage16(qg + j * 8, Qs + j * 8);
  for (int j = threadIdx.x; j < DCH * DCH / 8;  j += 256) stage16(cg + j * 8, Cs + j * 8);
  stage_fence();
  int t = tbh >> 6, bb = (tbh >> 3) & 7, h = tbh & 7;
  int lane = threadIdx.x & 31, wave = threadIdx.x >> 5;
  int m = lane & 15;
  for (int it = 0; it < 16; ++it) {       // 8 waves x 16 tiles = 128 (16x16) tiles
    int tile = wave * 16 + it;
    int nt = tile >> 4, et = tile & 15;
    const _Float16* arow = Qs + (size_t)(nt * 16 + m) * DCH;   // A[m][k]=Q[n][d]
    const _Float16* bcol = Cs + (size_t)(et * 16 + m) * DCH;   // B[k][n]=kv[d][e]
    v8f acc = {};
    #pragma unroll
    for (int kc = 0; kc < 8; ++kc) {      // inner d = 256 = 8 * 32
      v16h a = frag_a(arow + kc * 32, lane);
      v16h b = frag_b(bcol + kc * 32, lane);
      acc = __builtin_amdgcn_wmma_f32_16x16x32_f16(false, a, false, b, (short)0, acc, false, false);
    }
    int hi = lane >> 4;
    int nrow0 = nt * 16 + hi * 8;
    int e = et * 16 + m;
    // out(t,b,d',n') with flat(d',n') = nseq*(H*D) + h*D + e
    size_t base = (size_t)t * PER_T + (size_t)bb * (DCH * NLEN) + (size_t)h * DCH + e;
    #pragma unroll
    for (int v = 0; v < 8; ++v)
      att[base + (size_t)(nrow0 + v) * (HEADS * DCH)] = acc[v];
  }
}

// ================================================================ host
extern "C" void kernel_launch(void* const* d_in, const int* in_sizes, int n_in,
                              void* d_out, int out_size, void* d_ws, size_t ws_size,
                              hipStream_t stream) {
  (void)in_sizes; (void)n_in; (void)out_size; (void)ws_size;
  const float* x      = (const float*)d_in[0];
  const float* plifw  = (const float*)d_in[1];
  const float* plifb  = (const float*)d_in[2];
  const float* dw_w   = (const float*)d_in[3];
  const float* dw_g   = (const float*)d_in[4];
  const float* dw_b   = (const float*)d_in[5];
  const float* q_w    = (const float*)d_in[6];
  const float* q_g    = (const float*)d_in[7];
  const float* q_b    = (const float*)d_in[8];
  const float* k_w    = (const float*)d_in[9];
  const float* k_g    = (const float*)d_in[10];
  const float* k_b    = (const float*)d_in[11];
  const float* v_w    = (const float*)d_in[12];
  const float* v_g    = (const float*)d_in[13];
  const float* v_b    = (const float*)d_in[14];
  const float* pr_w   = (const float*)d_in[15];
  const float* pr_g   = (const float*)d_in[16];
  const float* pr_b   = (const float*)d_in[17];
  float* out = (float*)d_out;

  char* ws = (char*)d_ws;
  size_t off = 0;
  auto alloc = [&](size_t bytes) -> char* {
    char* p = ws + off;
    off = (off + bytes + 255) & ~(size_t)255;
    return p;
  };
  float* wsm  = (float*)alloc(256 * 4);
  float* cosT = (float*)alloc(NSEQ * DCH * 4);
  float* sinT = (float*)alloc(NSEQ * DCH * 4);
  float* s1   = (float*)alloc(256 * 4);
  float* s2   = (float*)alloc(256 * 4);
  float* sc   = (float*)alloc(256 * 4);
  float* sh   = (float*)alloc(256 * 4);
  float* bufA = (float*)alloc((size_t)TOT * 4);            // spikes / att / att-spikes
  float* bufB = (float*)alloc((size_t)TOT * 4);            // conv outputs (reused)
  float* bufC = (float*)alloc((size_t)TOT * 4);            // xs ("flat") spikes
  _Float16* qf  = (_Float16*)alloc((size_t)TOT * 2);
  _Float16* kf  = (_Float16*)alloc((size_t)TOT * 2);
  _Float16* vf  = (_Float16*)alloc((size_t)TOT * 2);
  _Float16* kvf = (_Float16*)alloc((size_t)TS * BBATCH * HEADS * DCH * DCH * 2);

  const int GP = PER_T / 256;        // 8192
  const int GPAIR = PER_T / 512;     // 4096
  const int GT = TOT / 256;          // 32768
  const int GCONV = (TS * BBATCH) * DCH;  // 8192 (one block per (m,d) row)
  const float cnt = (float)((TS * BBATCH) * NLEN);

  // tables
  k_softmax<<<1, 256, 0, stream>>>(plifw, wsm);
  k_rope_tables<<<NSEQ, 256, 0, stream>>>(cosT, sinT);

  // head_lif (PLIF) -> spikes in bufA
  k_plif<<<GP, 256, 0, stream>>>(x, wsm, plifb, bufA);

  // dwconv(5) + BN + LIF -> xs (bufC)
  (void)hipMemsetAsync(s1, 0, 256 * 4, stream);
  (void)hipMemsetAsync(s2, 0, 256 * 4, stream);
  k_dwconv_stats<<<GCONV, 256, 0, stream>>>(bufA, dw_w, 5, bufB, s1, s2);
  k_bn_finalize<<<1, 256, 0, stream>>>(s1, s2, dw_g, dw_b, cnt, sc, sh);
  k_bn_lif<<<GP, 256, 0, stream>>>(bufB, sc, sh, bufC);

  // q branch: conv(3)+BN+LIF+rope -> qf (f16, [tbh][n][d])
  (void)hipMemsetAsync(s1, 0, 256 * 4, stream);
  (void)hipMemsetAsync(s2, 0, 256 * 4, stream);
  k_dwconv_stats<<<GCONV, 256, 0, stream>>>(bufC, q_w, 3, bufB, s1, s2);
  k_bn_finalize<<<1, 256, 0, stream>>>(s1, s2, q_g, q_b, cnt, sc, sh);
  k_bn_lif_rope_f16<<<GPAIR, 256, 0, stream>>>(bufB, sc, sh, cosT, sinT, 1, 0, qf);

  // k branch -> kf (f16, transposed [tbh][d][n])
  (void)hipMemsetAsync(s1, 0, 256 * 4, stream);
  (void)hipMemsetAsync(s2, 0, 256 * 4, stream);
  k_dwconv_stats<<<GCONV, 256, 0, stream>>>(bufC, k_w, 3, bufB, s1, s2);
  k_bn_finalize<<<1, 256, 0, stream>>>(s1, s2, k_g, k_b, cnt, sc, sh);
  k_bn_lif_rope_f16<<<GPAIR, 256, 0, stream>>>(bufB, sc, sh, cosT, sinT, 1, 1, kf);

  // v branch (no rope) -> vf (f16, transposed [tbh][d][n])
  (void)hipMemsetAsync(s1, 0, 256 * 4, stream);
  (void)hipMemsetAsync(s2, 0, 256 * 4, stream);
  k_dwconv_stats<<<GCONV, 256, 0, stream>>>(bufC, v_w, 3, bufB, s1, s2);
  k_bn_finalize<<<1, 256, 0, stream>>>(s1, s2, v_g, v_b, cnt, sc, sh);
  k_bn_lif_rope_f16<<<GPAIR, 256, 0, stream>>>(bufB, sc, sh, cosT, sinT, 0, 1, vf);

  // attention: kvT = (K^T V)^T * scale ; att = Q @ kv  (f16 WMMA, one WG per (t,b,h))
  k_kv_wmma<<<TS * BBATCH * HEADS, 256, 2 * DCH * NSEQ * sizeof(_Float16), stream>>>(kf, vf, kvf);
  k_att_wmma<<<TS * BBATCH * HEADS, 256, (NSEQ * DCH + DCH * DCH) * sizeof(_Float16), stream>>>(qf, kvf, bufA);

  // attn_lif (in-place per-element scan is safe)
  k_lif<<<GP, 256, 0, stream>>>(bufA, bufA);

  // proj: conv(3) + BN -> out
  (void)hipMemsetAsync(s1, 0, 256 * 4, stream);
  (void)hipMemsetAsync(s2, 0, 256 * 4, stream);
  k_dwconv_stats<<<GCONV, 256, 0, stream>>>(bufA, pr_w, 3, bufB, s1, s2);
  k_bn_finalize<<<1, 256, 0, stream>>>(s1, s2, pr_g, pr_b, cnt, sc, sh);
  k_bn_apply<<<GT, 256, 0, stream>>>(bufB, sc, sh, out);
}
